// Video_multilevel_encoding_64957085385198
// MI455X (gfx1250) — matrix operand order repaired and checked by reading, not compiled, not run on hardware
//
#include <hip/hip_runtime.h>
#include <hip/hip_bf16.h>
#include <math.h>

#define B_    128
#define T_    80
#define D_    2048
#define H_    512
#define G3    1536     // 3*H
#define KCH   512      // conv out channels
#define MAPIN 5120
#define MAPOUT 2048

typedef __attribute__((ext_vector_type(16))) __bf16          v16bf;
typedef __attribute__((ext_vector_type(16))) unsigned short  v16u;
typedef __attribute__((ext_vector_type(8)))  unsigned short  v8u;
typedef __attribute__((ext_vector_type(4)))  unsigned short  v4u;
typedef __attribute__((ext_vector_type(4)))  float           v4f;
typedef __attribute__((ext_vector_type(8)))  float           v8f;

// native f32 -> bf16 (RTE) convert; lowers to v_cvt (pairs fold into v_cvt_pk_bf16_f32)
static __device__ __forceinline__ unsigned short bf16u(float f) {
  __bf16 h = (__bf16)f;
  return __builtin_bit_cast(unsigned short, h);
}

static __device__ __forceinline__ v8f wmma_bf16(v16bf a, v16bf b, v8f c) {
  return __builtin_amdgcn_wmma_f32_16x16x32_bf16(false, a, false, b, (short)0, c, false, false);
}

// A fragment (16x32 bf16) from an LDS tile [m][k], ld multiple of 8.
// Lane L (m=L&15, g=L>>4) holds two contiguous 8-element runs: k = g*8+e and k = 16+g*8+e.
static __device__ __forceinline__ v16bf frag_a_lds(const unsigned short* tile, int ld, int lane) {
  const int m = lane & 15, g = lane >> 4;
  const unsigned short* p = tile + m * ld + g * 8;
  v8u lo = *(const v8u*)p;          // ds_load_b128
  v8u hi = *(const v8u*)(p + 16);   // ds_load_b128
  v16u t = __builtin_shufflevector(lo, hi, 0,1,2,3,4,5,6,7,8,9,10,11,12,13,14,15);
  return __builtin_bit_cast(v16bf, t);
}

// B fragment from fragment-linear packed weights: frag f stores 32 lanes x 16 u16 contiguous.
static __device__ __forceinline__ v16bf frag_b_pack(const unsigned short* pack, long long f, int lane) {
  v16u t = *(const v16u*)(pack + ((size_t)f * 32 + lane) * 16);  // 2x global_load_b128
  return __builtin_bit_cast(v16bf, t);
}

// Async global->LDS 16B copy (CDNA5 GLOBAL_LOAD_ASYNC_TO_LDS_B128, tracked by ASYNCcnt).
// ldsoff = AS(3) byte offset (low 32 bits of flat shared pointer); gsrc = global address.
static __device__ __forceinline__ void async_g2l_b128(unsigned ldsoff, const void* gsrc) {
  asm volatile("global_load_async_to_lds_b128 %0, %1, off"
               :: "v"(ldsoff), "v"(gsrc) : "memory");
}
static __device__ __forceinline__ void wait_async0() {
  asm volatile("s_wait_asynccnt 0x0" ::: "memory");
}

// ------------------------------------------------- pack W[N][K] -> B-fragment-linear bf16
// dst[((kc*NT + nt)*32 + L)*16 + e] = bf16(W[nt*16 + (L&15)][kc*32 + (e>>3)*16 + (L>>4)*8 + (e&7)])
__global__ void k_pack_b(const float* __restrict__ src, unsigned short* __restrict__ dst,
                         int N, int K) {
  long long i = (long long)blockIdx.x * 256 + threadIdx.x;
  if (i >= (long long)N * K) return;
  const int e = (int)(i & 15);
  const int L = (int)((i >> 4) & 31);
  const long long f = i >> 9;
  const int NT = N >> 4;
  const int nt = (int)(f % NT), kc = (int)(f / NT);
  const int n = nt * 16 + (L & 15);
  const int k = kc * 32 + ((e >> 3) << 4) + ((L >> 4) << 3) + (e & 7);
  dst[i] = bf16u(src[(size_t)n * K + k]);
}

// ------------------------------------------------- pre = videos @ Wih^T + bih
// M=10240, N=2x1536, K=2048. Block tile 128x64, 8 waves -> 32x32 each. B direct from pack.
__global__ __launch_bounds__(256) void k_gemm_pre(
    const float* __restrict__ videos,          // [10240][2048]
    const unsigned short* __restrict__ WihP,   // packed, [2][64*96 frags]
    const float* __restrict__ bih_f, const float* __restrict__ bih_b,
    float* __restrict__ pre)                   // [2][10240][1536]
{
  const int tid = threadIdx.x, lane = tid & 31, wid = tid >> 5;
  const int m0 = blockIdx.x * 128;
  const int ng = blockIdx.y * 64;
  const int dir = (ng >= G3) ? 1 : 0;
  const int n0 = ng - dir * G3;
  const unsigned short* Bp = WihP + (size_t)dir * D_ * G3;
  const float* bih = dir ? bih_b : bih_f;

  __shared__ unsigned short As[128][40];

  const int wm = (wid >> 1) * 32;
  const int wn = (wid & 1) * 32;
  const int nt0 = (n0 + wn) >> 4;              // n-tile index of first accumulator column
  v8f a00 = {}, a01 = {}, a10 = {}, a11 = {};

  for (int kc = 0; kc < D_ / 32; ++kc) {
    const int k0 = kc * 32;
#pragma unroll
    for (int i = 0; i < 4; ++i) {              // 128x32 A: float4 reads, 4xu16 stores
      int q = i * 256 + tid;                   // 0..1023
      int r = q >> 3, c4 = (q & 7) * 4;
      v4f x = *(const v4f*)(videos + (size_t)(m0 + r) * D_ + k0 + c4);
      v4u y; y[0] = bf16u(x[0]); y[1] = bf16u(x[1]); y[2] = bf16u(x[2]); y[3] = bf16u(x[3]);
      *(v4u*)&As[r][c4] = y;
    }
    __syncthreads();
    v16bf fa0 = frag_a_lds(&As[wm][0], 40, lane);
    v16bf fa1 = frag_a_lds(&As[wm + 16][0], 40, lane);
    v16bf fb0 = frag_b_pack(Bp, (long long)kc * 96 + nt0,     lane);
    v16bf fb1 = frag_b_pack(Bp, (long long)kc * 96 + nt0 + 1, lane);
    a00 = wmma_bf16(fa0, fb0, a00);
    a01 = wmma_bf16(fa0, fb1, a01);
    a10 = wmma_bf16(fa1, fb0, a10);
    a11 = wmma_bf16(fa1, fb1, a11);
    __syncthreads();
  }
  const int g = lane >> 4, nl = lane & 15;
#pragma unroll
  for (int j = 0; j < 8; ++j) {
    int mA = m0 + wm + j + g * 8, mB = mA + 16;
    int nA = n0 + wn + nl, nB = nA + 16;
    pre[((size_t)dir * (B_ * T_) + mA) * G3 + nA] = a00[j] + bih[nA];
    pre[((size_t)dir * (B_ * T_) + mA) * G3 + nB] = a01[j] + bih[nB];
    pre[((size_t)dir * (B_ * T_) + mB) * G3 + nA] = a10[j] + bih[nA];
    pre[((size_t)dir * (B_ * T_) + mB) * G3 + nB] = a11[j] + bih[nB];
  }
}

// ------------------------------------------------- persistent GRU recurrence + fused mean
// grid (B/16, 2 dirs). Hidden state AND the masked temporal mean live in REGISTERS
// (each (m,col) owned by one thread for all 80 steps); LDS holds only the bf16 h copy
// feeding WMMA A-fragments. Mean is written as bf16 straight into xfeat[:,0:1024].
__global__ __launch_bounds__(256) void k_gru(
    const float* __restrict__ pre,             // [2][B*T][1536]
    const unsigned short* __restrict__ WhhP,   // packed, [2][16*96 frags]
    const float* __restrict__ bhh_f, const float* __restrict__ bhh_b,
    const float* __restrict__ mask,            // [B][T]
    const int* __restrict__ lengths,           // [B]
    unsigned short* __restrict__ h2m,          // [B][T][1024] bf16 masked gru output
    unsigned short* __restrict__ xfeat)        // [B][5120] bf16, cols 0..1023 = mean
{
  const int tid = threadIdx.x, lane = tid & 31, wid = tid >> 5;
  const int dir = blockIdx.y;
  const int b0 = blockIdx.x * 16;
  const unsigned short* Bp = WhhP + (size_t)dir * H_ * G3;
  const float* bhh = dir ? bhh_b : bhh_f;

  __shared__ unsigned short h_b[16][H_ + 8];   // bf16 h, ld=520 (8-aligned)
  for (int i = tid; i < 16 * (H_ + 8); i += 256) ((unsigned short*)h_b)[i] = 0;
  __syncthreads();

  const int g = lane >> 4, nl = lane & 15;
  const v8f vzero = {};
  float hreg[4][8], macc[4][8];
#pragma unroll
  for (int ti = 0; ti < 4; ++ti)
#pragma unroll
    for (int j = 0; j < 8; ++j) { hreg[ti][j] = 0.f; macc[ti][j] = 0.f; }

  // loop-invariant per-thread bases (hoisted address math)
  const float* prowBase[8];
  int obase[8];
#pragma unroll
  for (int j = 0; j < 8; ++j) {
    const int b = b0 + j + g * 8;
    prowBase[j] = pre + ((size_t)dir * (B_ * T_) + (size_t)b * T_) * G3;
    obase[j] = b * T_ * 1024 + dir * 512;
  }

  for (int s = 0; s < T_; ++s) {
    const int t = dir ? (T_ - 1 - s) : s;
    v8f ar[4], az[4], an[4];
#pragma unroll
    for (int ti = 0; ti < 4; ++ti) { ar[ti] = vzero; az[ti] = vzero; an[ti] = vzero; }

    for (int kc = 0; kc < H_ / 32; ++kc) {     // gh = h @ Whh^T
      v16bf fa = frag_a_lds(&h_b[0][kc * 32], H_ + 8, lane);
#pragma unroll
      for (int ti = 0; ti < 4; ++ti) {
        const int q = wid * 4 + ti;            // column group 0..31
        v16bf br = frag_b_pack(Bp, (long long)kc * 96 + q,      lane);
        v16bf bz = frag_b_pack(Bp, (long long)kc * 96 + 32 + q, lane);
        v16bf bn = frag_b_pack(Bp, (long long)kc * 96 + 64 + q, lane);
        ar[ti] = wmma_bf16(fa, br, ar[ti]);
        az[ti] = wmma_bf16(fa, bz, az[ti]);
        an[ti] = wmma_bf16(fa, bn, an[ti]);
      }
    }
    __syncthreads();                           // all A-fragment reads of h_b complete

    const size_t tG3 = (size_t)t * G3;
    const int t1024 = t * 1024;
    float mv[8];
#pragma unroll
    for (int j = 0; j < 8; ++j) mv[j] = mask[(b0 + j + g * 8) * T_ + t];

#pragma unroll
    for (int ti = 0; ti < 4; ++ti) {
      const int col = (wid * 4 + ti) * 16 + nl;
#pragma unroll
      for (int j = 0; j < 8; ++j) {
        const float* prow = prowBase[j] + tG3;
        float gr = ar[ti][j] + bhh[col];
        float gz = az[ti][j] + bhh[col + 512];
        float gn = an[ti][j] + bhh[col + 1024];
        float r  = 1.f / (1.f + expf(-(prow[col] + gr)));
        float z  = 1.f / (1.f + expf(-(prow[col + 512] + gz)));
        float nn = tanhf(prow[col + 1024] + r * gn);
        float hn = (1.f - z) * nn + z * hreg[ti][j];
        hreg[ti][j] = hn;
        h_b[j + g * 8][col] = bf16u(hn);
        float hm = hn * mv[j];
        macc[ti][j] += hm;
        h2m[obase[j] + t1024 + col] = bf16u(hm);
      }
    }
    __syncthreads();
  }

  // fused masked temporal mean -> xfeat[:, dir*512 + col] (bf16)
#pragma unroll
  for (int ti = 0; ti < 4; ++ti) {
    const int col = (wid * 4 + ti) * 16 + nl;
#pragma unroll
    for (int j = 0; j < 8; ++j) {
      const int b = b0 + j + g * 8;
      xfeat[(size_t)b * MAPIN + dir * 512 + col] = bf16u(macc[ti][j] / (float)lengths[b]);
    }
  }
}

// ------------------------------------------------- conv-as-GEMM + bias + relu + max over time
// grid (B/16, KCH/128, 4 widths). Block tile 16x128, 8 waves -> one 16x16 tile each.
// Full 16x1024 activation slab staged per (p, tap) via ASYNC global->LDS copies
// (GLOBAL_LOAD_ASYNC_TO_LDS_B128, ASYNCcnt); inner k-loop is barrier-free.
// Pooled result written as bf16 straight into xfeat[:, 1024:3072].
__global__ __launch_bounds__(256) void k_conv(
    const unsigned short* __restrict__ h2m,    // [B][T][1024] bf16 masked
    const unsigned short* __restrict__ convP,  // packed, concat w=2..5
    const float* __restrict__ cb2, const float* __restrict__ cb3,
    const float* __restrict__ cb4, const float* __restrict__ cb5,
    unsigned short* __restrict__ xfeat)        // [B][5120]
{
  const int tid = threadIdx.x, lane = tid & 31, wid = tid >> 5;
  const int widx = blockIdx.z;                 // 0..3 -> w = 2..5
  const int w = widx + 2;
  size_t woff = 0;
  for (int i = 0; i < widx; ++i) woff += (size_t)(i + 2) * 1024 * KCH;
  const unsigned short* Bp = convP + woff;
  const float* cb = (widx == 0) ? cb2 : (widx == 1) ? cb3 : (widx == 2) ? cb4 : cb5;
  const int m0 = blockIdx.x * 16;
  const int n0 = blockIdx.y * 128;
  const int nt = (n0 >> 4) + wid;              // this wave's n-tile (of 32)

  __shared__ unsigned short As[16][1032];      // 16 x 1024, ld=1032 (8-aligned), ~33KB
  const unsigned asBase = (unsigned)(unsigned long long)&As[0][0];  // AS(3) byte offset

  const int g = lane >> 4, nl = lane & 15;
  v8f mx = {};
  const int P = T_ + w - 1;
  for (int p = 0; p < P; ++p) {
    v8f acc = {};
    for (int j = 0; j < w; ++j) {              // block-uniform
      const int tt = p - (w - 1) + j;
      if (tt < 0 || tt >= T_) continue;        // zero padding -> whole slab contributes 0
#pragma unroll
      for (int i = 0; i < 8; ++i) {            // async stage 16x1024 slab, 16B per op
        int q = i * 256 + tid;                 // 0..2047
        int r = q >> 7, c8 = (q & 127) * 8;
        async_g2l_b128(asBase + (unsigned)(r * 1032 + c8) * 2,
                       h2m + ((size_t)(m0 + r) * T_ + tt) * 1024 + c8);
      }
      wait_async0();                           // this wave's async copies landed
      __syncthreads();                         // publish slab across waves
      for (int cc = 0; cc < 32; ++cc) {        // barrier-free inner K loop
        v16bf fa = frag_a_lds(&As[0][cc * 32], 1032, lane);
        v16bf fb = frag_b_pack(Bp, (long long)(j * 32 + cc) * 32 + nt, lane);
        acc = wmma_bf16(fa, fb, acc);
      }
      __syncthreads();
    }
#pragma unroll
    for (int j = 0; j < 8; ++j) {
      float v = acc[j] + cb[n0 + wid * 16 + nl];
      v = v > 0.f ? v : 0.f;
      mx[j] = v > mx[j] ? v : mx[j];
    }
  }
#pragma unroll
  for (int j = 0; j < 8; ++j) {
    int b = m0 + j + g * 8;
    int col = 1024 + widx * KCH + n0 + wid * 16 + nl;
    xfeat[(size_t)b * MAPIN + col] = bf16u(mx[j]);
  }
}

// ------------------------------------------------- videos_origin -> xfeat[:, 3072:5120] (bf16)
__global__ void k_vo(const float* __restrict__ vo, unsigned short* __restrict__ xfeat) {
  int i = blockIdx.x * 256 + threadIdx.x;      // B * 2048
  if (i >= B_ * 2048) return;
  int b = i >> 11, c = i & 2047;
  xfeat[(size_t)b * MAPIN + 3072 + c] = bf16u(vo[i]);
}

// ------------------------------------------------- map GEMM + bias + BN scale
// M=128, N=2048, K=5120. Block tile 32x128, 8 waves -> 16x32 each.
__global__ __launch_bounds__(256) void k_gemm_map(
    const unsigned short* __restrict__ xfeat,  // [128][5120]
    const unsigned short* __restrict__ mapP,   // packed, 160*128 frags
    const float* __restrict__ map_b, const float* __restrict__ gamma,
    const float* __restrict__ beta, float* __restrict__ feat) // [128][2048]
{
  const int tid = threadIdx.x, lane = tid & 31, wid = tid >> 5;
  const int m0 = blockIdx.x * 32;
  const int n0 = blockIdx.y * 128;
  const int wm = (wid >> 2) * 16;
  const int wn = (wid & 3) * 32;
  const int nt0 = ((n0 + wn) >> 4);

  __shared__ unsigned short As[32][40];

  v8f a0 = {}, a1 = {};
  for (int kc = 0; kc < MAPIN / 32; ++kc) {
    const int k0 = kc * 32;
    {                                          // stage 32x32: 4 u16 per thread
      int r = tid >> 3, c4 = (tid & 7) * 4;
      *(v4u*)&As[r][c4] = *(const v4u*)(xfeat + (size_t)(m0 + r) * MAPIN + k0 + c4);
    }
    __syncthreads();
    v16bf fa = frag_a_lds(&As[wm][0], 40, lane);
    v16bf b0 = frag_b_pack(mapP, (long long)kc * 128 + nt0,     lane);
    v16bf b1 = frag_b_pack(mapP, (long long)kc * 128 + nt0 + 1, lane);
    a0 = wmma_bf16(fa, b0, a0);
    a1 = wmma_bf16(fa, b1, a1);
    __syncthreads();
  }
  const float invs = 1.0f / sqrtf(1.0f + 1e-5f);
  const int g = lane >> 4, nl = lane & 15;
#pragma unroll
  for (int j = 0; j < 8; ++j) {
    int m = m0 + wm + j + g * 8;
    int nA = n0 + wn + nl, nB = nA + 16;
    float vA = a0[j] + map_b[nA];
    float vB = a1[j] + map_b[nB];
    feat[(size_t)m * MAPOUT + nA] = vA * invs * gamma[nA] + beta[nA];
    feat[(size_t)m * MAPOUT + nB] = vB * invs * gamma[nB] + beta[nB];
  }
}

// ------------------------------------------------- row L2 normalization
__global__ __launch_bounds__(256) void k_l2norm(const float* __restrict__ feat,
                                                float* __restrict__ out) {
  const int b = blockIdx.x;
  __shared__ float red[256];
  float s = 0.f;
  for (int c = threadIdx.x; c < MAPOUT; c += 256) {
    float v = feat[(size_t)b * MAPOUT + c];
    s += v * v;
  }
  red[threadIdx.x] = s;
  __syncthreads();
  for (int st = 128; st > 0; st >>= 1) {
    if (threadIdx.x < st) red[threadIdx.x] += red[threadIdx.x + st];
    __syncthreads();
  }
  const float rn = 1.0f / sqrtf(red[0]);
  for (int c = threadIdx.x; c < MAPOUT; c += 256)
    out[(size_t)b * MAPOUT + c] = feat[(size_t)b * MAPOUT + c] * rn;
}

// ------------------------------------------------- host
extern "C" void kernel_launch(void* const* d_in, const int* in_sizes, int n_in,
                              void* d_out, int out_size, void* d_ws, size_t ws_size,
                              hipStream_t stream) {
  const float* videos  = (const float*)d_in[0];
  const float* vorigin = (const float*)d_in[1];
  const int*   lengths = (const int*)d_in[2];
  const float* mask    = (const float*)d_in[3];
  // d_in[4] = gru_text_out (unused by reference output)
  const float* Wih_f = (const float*)d_in[5];
  const float* Whh_f = (const float*)d_in[6];
  const float* bih_f = (const float*)d_in[7];
  const float* bhh_f = (const float*)d_in[8];
  const float* Wih_b = (const float*)d_in[9];
  const float* Whh_b = (const float*)d_in[10];
  const float* bih_b = (const float*)d_in[11];
  const float* bhh_b = (const float*)d_in[12];
  // d_in[13..16] = attention params (softmax over singleton axis -> identity; skipped)
  const float* cw2 = (const float*)d_in[17]; const float* cb2 = (const float*)d_in[18];
  const float* cw3 = (const float*)d_in[19]; const float* cb3 = (const float*)d_in[20];
  const float* cw4 = (const float*)d_in[21]; const float* cb4 = (const float*)d_in[22];
  const float* cw5 = (const float*)d_in[23]; const float* cb5 = (const float*)d_in[24];
  const float* map_W = (const float*)d_in[25]; const float* map_b = (const float*)d_in[26];
  const float* gamma = (const float*)d_in[27]; const float* beta = (const float*)d_in[28];

  char* ws = (char*)d_ws;
  size_t off = 0;
  auto alloc = [&](size_t bytes) -> void* {
    void* p = ws + off;
    off = (off + bytes + 255) & ~(size_t)255;
    return p;
  };
  float* pre = (float*)alloc(sizeof(float) * 2ULL * B_ * T_ * G3);           // 125.8 MB
  unsigned short* h2m  = (unsigned short*)alloc(2ULL * B_ * T_ * 1024);      //  21.0 MB
  unsigned short* WihP = (unsigned short*)alloc(2ULL * 2 * D_ * G3);         //  12.6 MB
  unsigned short* WhhP = (unsigned short*)alloc(2ULL * 2 * H_ * G3);         //   3.1 MB
  unsigned short* convP = (unsigned short*)alloc(2ULL * 14 * 1024 * KCH);    //  14.7 MB
  unsigned short* mapP  = (unsigned short*)alloc(2ULL * MAPIN * MAPOUT);     //  21.0 MB
  unsigned short* xfeat = (unsigned short*)alloc(2ULL * B_ * MAPIN);
  float* feat  = (float*)alloc(sizeof(float) * (size_t)B_ * MAPOUT);
  (void)ws_size; (void)in_sizes; (void)n_in; (void)out_size;

  auto tg = [](long long n) { return (unsigned)((n + 255) / 256); };
  // weight -> fragment-linear bf16 pack (B operands, one-time per call)
  k_pack_b<<<tg(1536LL * 2048), 256, 0, stream>>>(Wih_f, WihP,                    G3, D_);
  k_pack_b<<<tg(1536LL * 2048), 256, 0, stream>>>(Wih_b, WihP + (size_t)D_ * G3,  G3, D_);
  k_pack_b<<<tg(1536LL * 512),  256, 0, stream>>>(Whh_f, WhhP,                    G3, H_);
  k_pack_b<<<tg(1536LL * 512),  256, 0, stream>>>(Whh_b, WhhP + (size_t)H_ * G3,  G3, H_);
  k_pack_b<<<tg(512LL * 2 * 1024), 256, 0, stream>>>(cw2, convP,                     KCH, 2 * 1024);
  k_pack_b<<<tg(512LL * 3 * 1024), 256, 0, stream>>>(cw3, convP + 2ULL * 1024 * KCH, KCH, 3 * 1024);
  k_pack_b<<<tg(512LL * 4 * 1024), 256, 0, stream>>>(cw4, convP + 5ULL * 1024 * KCH, KCH, 4 * 1024);
  k_pack_b<<<tg(512LL * 5 * 1024), 256, 0, stream>>>(cw5, convP + 9ULL * 1024 * KCH, KCH, 5 * 1024);
  k_pack_b<<<tg((long long)MAPOUT * MAPIN), 256, 0, stream>>>(map_W, mapP, MAPOUT, MAPIN);

  dim3 gpre(80, 48);
  k_gemm_pre<<<gpre, 256, 0, stream>>>(videos, WihP, bih_f, bih_b, pre);

  dim3 ggru(B_ / 16, 2);
  k_gru<<<ggru, 256, 0, stream>>>(pre, WhhP, bhh_f, bhh_b, mask, lengths, h2m, xfeat);

  dim3 gconv(B_ / 16, KCH / 128, 4);
  k_conv<<<gconv, 256, 0, stream>>>(h2m, convP, cb2, cb3, cb4, cb5, xfeat);

  k_vo<<<(B_ * 2048) / 256, 256, 0, stream>>>(vorigin, xfeat);

  dim3 gmap(B_ / 32, MAPOUT / 128);
  k_gemm_map<<<gmap, 256, 0, stream>>>(xfeat, mapP, map_b, gamma, beta, feat);

  k_l2norm<<<B_, 256, 0, stream>>>(feat, (float*)d_out);
}
